// LinearNet_18622978196102
// MI455X (gfx1250) — compile-verified, matrix operand order checked by reading, not run on hardware
//
#include <hip/hip_runtime.h>
#include <stdint.h>

#define NDIM 4096
#define BK 32
#define NSTEP (NDIM / BK)
#define LDT 40   // padded LDS row stride in bf16 elements (80B, 16B-aligned chunks)

typedef __bf16 bf16;
typedef __attribute__((ext_vector_type(16))) __bf16 v16bf;
typedef __attribute__((ext_vector_type(8)))  __bf16 v8bf;
typedef __attribute__((ext_vector_type(4)))  __bf16 v4bf;
typedef __attribute__((ext_vector_type(8)))  float  v8f;
typedef __attribute__((ext_vector_type(4)))  float  v4f;

// ---- CDNA5 async global->LDS copy (ASYNCcnt path), guarded ----
#if defined(__has_builtin)
#if __has_builtin(__builtin_amdgcn_global_load_async_to_lds_b128)
#define HAVE_ASYNC_LDS 1
#endif
#endif

#ifdef HAVE_ASYNC_LDS
typedef int v4i __attribute__((vector_size(16)));
typedef __attribute__((address_space(1))) v4i as1_v4i;
typedef __attribute__((address_space(3))) v4i as3_v4i;
static __device__ inline void async_cp16(const bf16* g, bf16* l) {
    // generic LDS pointer: low 32 bits are the LDS byte offset (ISA aperture rule)
    __builtin_amdgcn_global_load_async_to_lds_b128(
        (as1_v4i*)(uintptr_t)g, (as3_v4i*)(uint32_t)(uintptr_t)l, 0, 0);
}
#if __has_builtin(__builtin_amdgcn_s_wait_asynccnt)
#define ASYNC_WAIT() __builtin_amdgcn_s_wait_asynccnt(0)
#else
#define ASYNC_WAIT() asm volatile("s_wait_asynccnt 0x0" ::: "memory")
#endif
#endif

static __device__ inline v8f wmma_bf16(v16bf a, v16bf b, v8f c) {
    return __builtin_amdgcn_wmma_f32_16x16x32_bf16(false, a, false, b, (short)0, c,
                                                   false, false);
}

// Fragment load from an LDS tile stored [row][k] (row = M for A, N for B).
// CDNA5 16-bit A/B layout: lane half 0 -> K {0..7,16..23}, half 1 -> K {8..15,24..31}.
static __device__ inline v16bf load_frag(const bf16* s, int row, int half) {
    v8bf lo = *(const v8bf*)(s + row * LDT + half * 8);
    v8bf hi = *(const v8bf*)(s + row * LDT + 16 + half * 8);
    return __builtin_shufflevector(lo, hi, 0, 1, 2, 3, 4, 5, 6, 7,
                                   8, 9, 10, 11, 12, 13, 14, 15);
}

// ---------- prep: elementwise f32 -> bf16 for x, W_r, W_i ----------
__global__ __launch_bounds__(256) void cvt_kernel(
    const float* __restrict__ x, const float* __restrict__ wr,
    const float* __restrict__ wi, bf16* __restrict__ xh,
    bf16* __restrict__ wrh, bf16* __restrict__ wih) {
    size_t i = ((size_t)blockIdx.x * 256 + threadIdx.x) * 4;
    v4f a = *(const v4f*)(x + i);
    v4f b = *(const v4f*)(wr + i);
    v4f c = *(const v4f*)(wi + i);
    v4bf ah, bh, ch;
#pragma unroll
    for (int r = 0; r < 4; ++r) {
        ah[r] = (bf16)a[r]; bh[r] = (bf16)b[r]; ch[r] = (bf16)c[r];
    }
    *(v4bf*)(xh + i) = ah;
    *(v4bf*)(wrh + i) = bh;
    *(v4bf*)(wih + i) = ch;
}

// ---------- prep: W^T in bf16 (wrt[m][k]=W_r[k][m], wit, wst=wrt+wit) ----------
__global__ __launch_bounds__(256) void transpose_kernel(
    const float* __restrict__ wr, const float* __restrict__ wi,
    bf16* __restrict__ wrt, bf16* __restrict__ wit, bf16* __restrict__ wst) {
    __shared__ float sR[32][33];
    __shared__ float sI[32][33];
    const int tx = threadIdx.x & 31, ty = threadIdx.x >> 5;
    const int k0 = blockIdx.y * 32, m0 = blockIdx.x * 32;
#pragma unroll
    for (int i = 0; i < 4; ++i) {
        int row = ty + i * 8;
        sR[row][tx] = wr[(size_t)(k0 + row) * NDIM + m0 + tx];
        sI[row][tx] = wi[(size_t)(k0 + row) * NDIM + m0 + tx];
    }
    __syncthreads();
#pragma unroll
    for (int i = 0; i < 4; ++i) {
        int row = ty + i * 8;
        float r = sR[tx][row];
        float m = sI[tx][row];
        size_t o = (size_t)(m0 + row) * NDIM + k0 + tx;
        wrt[o] = (bf16)r;
        wit[o] = (bf16)m;
        wst[o] = (bf16)(r + m);
    }
}

// ---------- stage 1: R_r = x W_r^T, R_i = x W_i^T (shared A tile) ----------
// Outputs stored TRANSPOSED (Rt[n][m]) in bf16, plus Rts = (R_r+R_i)^T.
__global__ __launch_bounds__(256) void gemm_stage1(
    const bf16* __restrict__ A, const bf16* __restrict__ B0,
    const bf16* __restrict__ B1, bf16* __restrict__ Ct0,
    bf16* __restrict__ Ct1, bf16* __restrict__ Cts) {
    __shared__ bf16 sA[2][128 * LDT];
    __shared__ bf16 sB0[2][128 * LDT];
    __shared__ bf16 sB1[2][128 * LDT];

    const int tid = threadIdx.x;
    const int lane = tid & 31, wave = tid >> 5;
    const int waveM = wave >> 2, waveN = wave & 3;  // 2 x 4 waves
    const int m0 = blockIdx.y * 128, n0 = blockIdx.x * 128;
    const int half = lane >> 4, l15 = lane & 15;

    // per-thread copy coordinates: 512 16B chunks per tile, 2 per thread
    const int r0 = tid >> 2, r1 = (tid + 256) >> 2;
    const int lcol = (tid & 3) * 8;
    const int lo0 = r0 * LDT + lcol, lo1 = r1 * LDT + lcol;

    v8f zero = {0.f, 0.f, 0.f, 0.f, 0.f, 0.f, 0.f, 0.f};
    v8f acc0[4][2], acc1[4][2];
#pragma unroll
    for (int ti = 0; ti < 4; ++ti)
#pragma unroll
        for (int tj = 0; tj < 2; ++tj) { acc0[ti][tj] = zero; acc1[ti][tj] = zero; }

#ifdef HAVE_ASYNC_LDS
    auto fetch = [&](int b, int k0) {
        async_cp16(A  + (size_t)(m0 + r0) * NDIM + k0 + lcol, &sA[b][lo0]);
        async_cp16(A  + (size_t)(m0 + r1) * NDIM + k0 + lcol, &sA[b][lo1]);
        async_cp16(B0 + (size_t)(n0 + r0) * NDIM + k0 + lcol, &sB0[b][lo0]);
        async_cp16(B0 + (size_t)(n0 + r1) * NDIM + k0 + lcol, &sB0[b][lo1]);
        async_cp16(B1 + (size_t)(n0 + r0) * NDIM + k0 + lcol, &sB1[b][lo0]);
        async_cp16(B1 + (size_t)(n0 + r1) * NDIM + k0 + lcol, &sB1[b][lo1]);
    };
#else
    v8bf rA0, rA1, rB00, rB01, rB10, rB11;
    auto fetch_regs = [&](int k0) {
        rA0  = *(const v8bf*)(A  + (size_t)(m0 + r0) * NDIM + k0 + lcol);
        rA1  = *(const v8bf*)(A  + (size_t)(m0 + r1) * NDIM + k0 + lcol);
        rB00 = *(const v8bf*)(B0 + (size_t)(n0 + r0) * NDIM + k0 + lcol);
        rB01 = *(const v8bf*)(B0 + (size_t)(n0 + r1) * NDIM + k0 + lcol);
        rB10 = *(const v8bf*)(B1 + (size_t)(n0 + r0) * NDIM + k0 + lcol);
        rB11 = *(const v8bf*)(B1 + (size_t)(n0 + r1) * NDIM + k0 + lcol);
    };
    auto stash = [&](int b) {
        *(v8bf*)(&sA[b][lo0])  = rA0;
        *(v8bf*)(&sA[b][lo1])  = rA1;
        *(v8bf*)(&sB0[b][lo0]) = rB00;
        *(v8bf*)(&sB0[b][lo1]) = rB01;
        *(v8bf*)(&sB1[b][lo0]) = rB10;
        *(v8bf*)(&sB1[b][lo1]) = rB11;
    };
#endif

    auto compute = [&](int b) {
        v16bf af[4], b0f[2], b1f[2];
#pragma unroll
        for (int ti = 0; ti < 4; ++ti)
            af[ti] = load_frag(sA[b], waveM * 64 + ti * 16 + l15, half);
#pragma unroll
        for (int tj = 0; tj < 2; ++tj) {
            b0f[tj] = load_frag(sB0[b], waveN * 32 + tj * 16 + l15, half);
            b1f[tj] = load_frag(sB1[b], waveN * 32 + tj * 16 + l15, half);
        }
#pragma unroll
        for (int ti = 0; ti < 4; ++ti)
#pragma unroll
            for (int tj = 0; tj < 2; ++tj) {
                acc0[ti][tj] = wmma_bf16(af[ti], b0f[tj], acc0[ti][tj]);
                acc1[ti][tj] = wmma_bf16(af[ti], b1f[tj], acc1[ti][tj]);
            }
    };

#ifdef HAVE_ASYNC_LDS
    fetch(0, 0);
    ASYNC_WAIT();
    __syncthreads();
    for (int k = 0; k < NSTEP - 1; ++k) {
        fetch((k + 1) & 1, (k + 1) * BK);
        compute(k & 1);
        ASYNC_WAIT();
        __syncthreads();
    }
    compute((NSTEP - 1) & 1);
#else
    fetch_regs(0);
    stash(0);
    __syncthreads();
    for (int k = 0; k < NSTEP - 1; ++k) {
        fetch_regs((k + 1) * BK);
        compute(k & 1);
        stash((k + 1) & 1);
        __syncthreads();
    }
    compute((NSTEP - 1) & 1);
#endif

    // Epilogue: C/D layout -> lane l15 = n, VGPR r = m (+8 for upper half).
#pragma unroll
    for (int ti = 0; ti < 4; ++ti)
#pragma unroll
        for (int tj = 0; tj < 2; ++tj) {
            int n = n0 + waveN * 32 + tj * 16 + l15;
            int mb = m0 + waveM * 64 + ti * 16 + half * 8;
            v8bf h0, h1, hs;
#pragma unroll
            for (int r = 0; r < 8; ++r) {
                float vr = acc0[ti][tj][r], vi = acc1[ti][tj][r];
                h0[r] = (bf16)vr;
                h1[r] = (bf16)vi;
                hs[r] = (bf16)(vr + vi);
            }
            size_t o = (size_t)n * NDIM + mb;
            *(v8bf*)(Ct0 + o) = h0;
            *(v8bf*)(Ct1 + o) = h1;
            *(v8bf*)(Cts + o) = hs;
        }
}

// ---------- stage 2: Karatsuba complex GEMM ----------
// P1 = Wr^T Rr, P2 = Wi^T Ri, P3 = (Wr+Wi)^T (Rr+Ri)
// C_r = P1 - P2 ; C_i = P3 - P1 - P2 ; out = [C_r | C_i]  (N x 2N, f32)
__global__ __launch_bounds__(256) void gemm_stage2(
    const bf16* __restrict__ A1, const bf16* __restrict__ A2,
    const bf16* __restrict__ A3,  // wrt, wit, wst : [m][k]
    const bf16* __restrict__ B1, const bf16* __restrict__ B2,
    const bf16* __restrict__ B3,  // rtr, rti, rts : [n][k]
    float* __restrict__ out) {
    __shared__ bf16 sA1[2][128 * LDT];
    __shared__ bf16 sA2[2][128 * LDT];
    __shared__ bf16 sA3[2][128 * LDT];
    __shared__ bf16 sB1[2][64 * LDT];
    __shared__ bf16 sB2[2][64 * LDT];
    __shared__ bf16 sB3[2][64 * LDT];

    const int tid = threadIdx.x;
    const int lane = tid & 31, wave = tid >> 5;
    const int waveM = wave >> 1, waveN = wave & 1;  // 4 x 2 waves
    const int m0 = blockIdx.y * 128, n0 = blockIdx.x * 64;
    const int half = lane >> 4, l15 = lane & 15;

    const int r0 = tid >> 2, r1 = (tid + 256) >> 2;
    const int lcol = (tid & 3) * 8;
    const int lo0 = r0 * LDT + lcol, lo1 = r1 * LDT + lcol;

    v8f zero = {0.f, 0.f, 0.f, 0.f, 0.f, 0.f, 0.f, 0.f};
    v8f p1[2][2], p2[2][2], p3[2][2];
#pragma unroll
    for (int ti = 0; ti < 2; ++ti)
#pragma unroll
        for (int tj = 0; tj < 2; ++tj) {
            p1[ti][tj] = zero; p2[ti][tj] = zero; p3[ti][tj] = zero;
        }

#ifdef HAVE_ASYNC_LDS
    auto fetch = [&](int b, int k0) {
        size_t ga0 = (size_t)(m0 + r0) * NDIM + k0 + lcol;
        size_t ga1 = (size_t)(m0 + r1) * NDIM + k0 + lcol;
        size_t gb  = (size_t)(n0 + r0) * NDIM + k0 + lcol;
        async_cp16(A1 + ga0, &sA1[b][lo0]);
        async_cp16(A1 + ga1, &sA1[b][lo1]);
        async_cp16(A2 + ga0, &sA2[b][lo0]);
        async_cp16(A2 + ga1, &sA2[b][lo1]);
        async_cp16(A3 + ga0, &sA3[b][lo0]);
        async_cp16(A3 + ga1, &sA3[b][lo1]);
        async_cp16(B1 + gb, &sB1[b][lo0]);
        async_cp16(B2 + gb, &sB2[b][lo0]);
        async_cp16(B3 + gb, &sB3[b][lo0]);
    };
#else
    v8bf ra10, ra11, ra20, ra21, ra30, ra31, rb1, rb2, rb3;
    auto fetch_regs = [&](int k0) {
        size_t ga0 = (size_t)(m0 + r0) * NDIM + k0 + lcol;
        size_t ga1 = (size_t)(m0 + r1) * NDIM + k0 + lcol;
        size_t gb  = (size_t)(n0 + r0) * NDIM + k0 + lcol;
        ra10 = *(const v8bf*)(A1 + ga0);
        ra11 = *(const v8bf*)(A1 + ga1);
        ra20 = *(const v8bf*)(A2 + ga0);
        ra21 = *(const v8bf*)(A2 + ga1);
        ra30 = *(const v8bf*)(A3 + ga0);
        ra31 = *(const v8bf*)(A3 + ga1);
        rb1 = *(const v8bf*)(B1 + gb);
        rb2 = *(const v8bf*)(B2 + gb);
        rb3 = *(const v8bf*)(B3 + gb);
    };
    auto stash = [&](int b) {
        *(v8bf*)(&sA1[b][lo0]) = ra10;
        *(v8bf*)(&sA1[b][lo1]) = ra11;
        *(v8bf*)(&sA2[b][lo0]) = ra20;
        *(v8bf*)(&sA2[b][lo1]) = ra21;
        *(v8bf*)(&sA3[b][lo0]) = ra30;
        *(v8bf*)(&sA3[b][lo1]) = ra31;
        *(v8bf*)(&sB1[b][lo0]) = rb1;
        *(v8bf*)(&sB2[b][lo0]) = rb2;
        *(v8bf*)(&sB3[b][lo0]) = rb3;
    };
#endif

    auto compute = [&](int b) {
        v16bf a1f[2], a2f[2], a3f[2], b1f[2], b2f[2], b3f[2];
#pragma unroll
        for (int ti = 0; ti < 2; ++ti) {
            int row = waveM * 32 + ti * 16 + l15;
            a1f[ti] = load_frag(sA1[b], row, half);
            a2f[ti] = load_frag(sA2[b], row, half);
            a3f[ti] = load_frag(sA3[b], row, half);
        }
#pragma unroll
        for (int tj = 0; tj < 2; ++tj) {
            int row = waveN * 32 + tj * 16 + l15;
            b1f[tj] = load_frag(sB1[b], row, half);
            b2f[tj] = load_frag(sB2[b], row, half);
            b3f[tj] = load_frag(sB3[b], row, half);
        }
#pragma unroll
        for (int ti = 0; ti < 2; ++ti)
#pragma unroll
            for (int tj = 0; tj < 2; ++tj) {
                p1[ti][tj] = wmma_bf16(a1f[ti], b1f[tj], p1[ti][tj]);
                p2[ti][tj] = wmma_bf16(a2f[ti], b2f[tj], p2[ti][tj]);
                p3[ti][tj] = wmma_bf16(a3f[ti], b3f[tj], p3[ti][tj]);
            }
    };

#ifdef HAVE_ASYNC_LDS
    fetch(0, 0);
    ASYNC_WAIT();
    __syncthreads();
    for (int k = 0; k < NSTEP - 1; ++k) {
        fetch((k + 1) & 1, (k + 1) * BK);
        compute(k & 1);
        ASYNC_WAIT();
        __syncthreads();
    }
    compute((NSTEP - 1) & 1);
#else
    fetch_regs(0);
    stash(0);
    __syncthreads();
    for (int k = 0; k < NSTEP - 1; ++k) {
        fetch_regs((k + 1) * BK);
        compute(k & 1);
        stash((k + 1) & 1);
        __syncthreads();
    }
    compute((NSTEP - 1) & 1);
#endif

#pragma unroll
    for (int ti = 0; ti < 2; ++ti)
#pragma unroll
        for (int tj = 0; tj < 2; ++tj) {
            int n = n0 + waveN * 32 + tj * 16 + l15;
            int mb = m0 + waveM * 32 + ti * 16 + half * 8;
#pragma unroll
            for (int r = 0; r < 8; ++r) {
                float a = p1[ti][tj][r], b = p2[ti][tj][r], c = p3[ti][tj][r];
                float cr = a - b;
                float ci = c - a - b;
                size_t o = (size_t)(mb + r) * (2 * NDIM) + n;
                out[o] = cr;
                out[o + NDIM] = ci;
            }
        }
}

extern "C" void kernel_launch(void* const* d_in, const int* in_sizes, int n_in,
                              void* d_out, int out_size, void* d_ws, size_t ws_size,
                              hipStream_t stream) {
    const float* x = (const float*)d_in[0];
    const float* Wr = (const float*)d_in[1];
    const float* Wi = (const float*)d_in[2];
    float* out = (float*)d_out;

    const size_t NN = (size_t)NDIM * NDIM;
    bf16* ws = (bf16*)d_ws;
    bf16* xh  = ws + 0 * NN;
    bf16* wrh = ws + 1 * NN;
    bf16* wih = ws + 2 * NN;
    bf16* wrt = ws + 3 * NN;
    bf16* wit = ws + 4 * NN;
    bf16* wst = ws + 5 * NN;
    bf16* rtr = ws + 6 * NN;
    bf16* rti = ws + 7 * NN;
    bf16* rts = ws + 8 * NN;

    cvt_kernel<<<(unsigned)(NN / (256 * 4)), 256, 0, stream>>>(x, Wr, Wi, xh, wrh, wih);
    transpose_kernel<<<dim3(NDIM / 32, NDIM / 32), 256, 0, stream>>>(Wr, Wi, wrt, wit, wst);
    gemm_stage1<<<dim3(NDIM / 128, NDIM / 128), 256, 0, stream>>>(xh, wrh, wih, rtr, rti, rts);
    gemm_stage2<<<dim3(NDIM / 64, NDIM / 128), 256, 0, stream>>>(wrt, wit, wst, rtr, rti, rts, out);
}
